// Top2Gate_6536940225051
// MI455X (gfx1250) — matrix operand order repaired and checked by reading, not compiled, not run on hardware
//
#include <hip/hip_runtime.h>
#include <hip/hip_bf16.h>
#include <stdint.h>

#define S_TOK 65536
#define DDIM  2048
#define NEXP  64
#define CAP   2048            // 2 * ceil(S/E) with CF=1.0
#define ROWS_PER_BLK 128
#define NBLK  (S_TOK / ROWS_PER_BLK)   // 512

typedef __attribute__((ext_vector_type(16))) __bf16 v16bf;
typedef __attribute__((ext_vector_type(8)))  float  v8f;
typedef __attribute__((__vector_size__(16))) int    v4i_t;

struct B32x2 { uint4 a, b; };          // 32 bytes, matches v16bf
struct A8    { unsigned int w[8]; };   // 32 bytes, matches v16bf

#if __has_builtin(__builtin_amdgcn_global_load_async_to_lds_b128)
#define HAVE_ASYNC_COPY 1
#else
#define HAVE_ASYNC_COPY 0
#endif

__device__ __forceinline__ unsigned short f2bf_rne(float f) {
    // round-to-nearest-even f32 -> bf16 (used once, in the pack kernel)
    unsigned int u = __float_as_uint(f);
    unsigned int r = u + 0x7FFFu + ((u >> 16) & 1u);
    return (unsigned short)(r >> 16);
}

__device__ __forceinline__ unsigned int pkbf16_trunc(float lo, float hi) {
    // pack two f32 into two bf16 by truncation (cheap: ~v_perm_b32)
    return (__float_as_uint(hi) & 0xFFFF0000u) | (__float_as_uint(lo) >> 16);
}

// ---------------------------------------------------------------------------
// K0: pack wg (E x D f32, row-major) into WMMA-B fragment order, bf16 (RNE).
// Flat layout: [((n*64 + kb)*32 + lane)*16 + j]  (ushort)
//   lane 0..15 : N = n*16+lane,    K = kb*32 + 0..15
//   lane 16..31: N = n*16+lane-16, K = kb*32 + 16..31
// Also zero me[64] (graph replay: must self-init accumulators every call).
// ---------------------------------------------------------------------------
__global__ void __launch_bounds__(256)
pack_wg_kernel(const float* __restrict__ wg, unsigned short* __restrict__ wgpack,
               float* __restrict__ me) {
    int gid  = blockIdx.x * 256 + threadIdx.x;   // 0 .. 131071
    int j    = gid & 15;
    int lane = (gid >> 4) & 31;
    int kb   = (gid >> 9) & 63;
    int n    = gid >> 15;
    int ncol = n * 16 + (lane & 15);
    int k    = kb * 32 + ((lane < 16) ? 0 : 16) + j;
    wgpack[gid] = f2bf_rne(wg[ncol * DDIM + k]);
    if (gid < NEXP) me[gid] = 0.f;
}

// ---------------------------------------------------------------------------
// K1: fused logits GEMM (bf16 WMMA) + per-row top2/softmax + block histograms
// 512 blocks x 256 threads (8 waves). Wave w handles rows blk*128 + w*16 .. +15.
// ---------------------------------------------------------------------------
__global__ void __launch_bounds__(256)
gemm_top2_kernel(const float* __restrict__ x, const unsigned short* __restrict__ wgpack,
                 float* __restrict__ g1, float* __restrict__ g2,
                 float* __restrict__ i1f, float* __restrict__ i2f,
                 int* __restrict__ idx1, int* __restrict__ idx2,
                 int* __restrict__ cnt1, int* __restrict__ cnt2,
                 float* __restrict__ me) {
    __shared__ unsigned short sB[4 * 16 * 32 * 16];   // 64 KB: one quarter of packed wg
    __shared__ float sLogits[8][16][NEXP];            // 32 KB
    __shared__ float sMe[NEXP];
    __shared__ int   sC1[NEXP], sC2[NEXP];

    const int tid  = threadIdx.x;
    const int lane = tid & 31;
    const int wave = tid >> 5;

    if (tid < NEXP) { sMe[tid] = 0.f; sC1[tid] = 0; sC2[tid] = 0; }

    const int row    = blockIdx.x * ROWS_PER_BLK + wave * 16 + (lane & 15);
    const int koff   = (lane < 16) ? 0 : 16;          // contiguous A-fragment layout (k = 16h + s), matches packed B
    const float* px  = x + (size_t)row * DDIM;

    v8f acc0 = {}, acc1 = {}, acc2 = {}, acc3 = {};

    for (int p = 0; p < 4; ++p) {
        __syncthreads();
        // Stage 64 KB of B for kb = p*16 .. p*16+15 (all 4 N-tiles).
        // local ushort idx: ((n*16 + kbl)*32 + lane)*16 + j
        for (int c = tid; c < 4096; c += 256) {        // 4096 x 16-byte chunks
            int o    = c * 8;                          // ushort offset
            int nn   = o >> 13;                        // / 8192
            int rem  = o & 8191;
            int kbl  = rem >> 9;                       // / 512
            int rem2 = rem & 511;
            int gsrc = (nn * 64 + p * 16 + kbl) * 512 + rem2;
#if HAVE_ASYNC_COPY
            __builtin_amdgcn_global_load_async_to_lds_b128(
                (__attribute__((address_space(1))) v4i_t*)&wgpack[gsrc],
                (__attribute__((address_space(3))) v4i_t*)&sB[o],
                0, 0);
#else
            *(uint4*)&sB[o] = *(const uint4*)&wgpack[gsrc];
#endif
        }
#if HAVE_ASYNC_COPY
#if __has_builtin(__builtin_amdgcn_s_wait_asynccnt)
        __builtin_amdgcn_s_wait_asynccnt(0);
#else
        asm volatile("s_wait_asynccnt 0x0" ::: "memory");
#endif
#endif
        __syncthreads();

        for (int kbl = 0; kbl < 16; ++kbl) {
            const int k0 = (p * 16 + kbl) * 32;
            // ---- A fragment: 16 f32 -> 16 bf16 per lane (fast truncation) ----
            const float4* pa = (const float4*)(px + k0 + koff);
            float4 f0 = pa[0], f1 = pa[1];
            const float4* pb = (const float4*)(px + k0 + koff + 8);
            float4 f2 = pb[0], f3 = pb[1];
            A8 aw;
            aw.w[0] = pkbf16_trunc(f0.x, f0.y);
            aw.w[1] = pkbf16_trunc(f0.z, f0.w);
            aw.w[2] = pkbf16_trunc(f1.x, f1.y);
            aw.w[3] = pkbf16_trunc(f1.z, f1.w);
            aw.w[4] = pkbf16_trunc(f2.x, f2.y);
            aw.w[5] = pkbf16_trunc(f2.z, f2.w);
            aw.w[6] = pkbf16_trunc(f3.x, f3.y);
            aw.w[7] = pkbf16_trunc(f3.z, f3.w);
            v16bf afrag = __builtin_bit_cast(v16bf, aw);

            // ---- B fragments from LDS + 4x WMMA (N tiles 0..3) ----
#define DO_TILE(NT, ACC)                                                        \
            {                                                                   \
                const uint4* bp = (const uint4*)&sB[(((NT)*16 + kbl)*32 + lane)*16]; \
                B32x2 t; t.a = bp[0]; t.b = bp[1];                              \
                v16bf bfrag = __builtin_bit_cast(v16bf, t);                     \
                ACC = __builtin_amdgcn_wmma_f32_16x16x32_bf16(                  \
                        false, afrag, false, bfrag, (short)0, ACC, false, false); \
            }
            DO_TILE(0, acc0)
            DO_TILE(1, acc1)
            DO_TILE(2, acc2)
            DO_TILE(3, acc3)
#undef DO_TILE
        }
    }

    // ---- spill 16x64 logit tile to LDS (C layout: VGPR r -> M = r + 8*(lane>=16), N = lane&15)
    const int colBase = lane & 15;
    const int rowHalf = (lane >> 4) * 8;
#pragma unroll
    for (int r = 0; r < 8; ++r) {
        sLogits[wave][rowHalf + r][ 0 + colBase] = acc0[r];
        sLogits[wave][rowHalf + r][16 + colBase] = acc1[r];
        sLogits[wave][rowHalf + r][32 + colBase] = acc2[r];
        sLogits[wave][rowHalf + r][48 + colBase] = acc3[r];
    }
    __syncthreads();

    // ---- one thread per row: top2 + softmax + histograms ----
    if (tid < ROWS_PER_BLK) {
        const float* lrow = &sLogits[tid >> 4][tid & 15][0];
        float m1 = -3.4e38f; int i1 = 0;
        float m2 = -3.4e38f; int i2 = 0;
        for (int e = 0; e < NEXP; ++e) {
            float v = lrow[e];
            if (v > m1)      { m2 = m1; i2 = i1; m1 = v; i1 = e; }
            else if (v > m2) { m2 = v;  i2 = e; }
        }
        float s = 0.f;
        for (int e = 0; e < NEXP; ++e) s += expf(lrow[e] - m1);
        float inv = 1.f / s;
        for (int e = 0; e < NEXP; ++e) atomicAdd(&sMe[e], expf(lrow[e] - m1) * inv);

        int r = blockIdx.x * ROWS_PER_BLK + tid;
        g1[r]  = inv;                          // exp(m1-m1)/s
        g2[r]  = expf(m2 - m1) * inv;
        i1f[r] = (float)i1;  i2f[r] = (float)i2;
        idx1[r] = i1;        idx2[r] = i2;
        atomicAdd(&sC1[i1], 1);
        atomicAdd(&sC2[i2], 1);
    }
    __syncthreads();
    if (tid < NEXP) {
        atomicAdd(&me[tid], sMe[tid]);
        cnt1[blockIdx.x * NEXP + tid] = sC1[tid];
        cnt2[blockIdx.x * NEXP + tid] = sC2[tid];
    }
}

// ---------------------------------------------------------------------------
// K2: per-expert exclusive scan over 512 block histograms; l_aux.
//   off2 has total1 folded in (locations2 = cumsum(mask2)-1 + sum(mask1)).
// ---------------------------------------------------------------------------
__global__ void __launch_bounds__(64)
scan_kernel(const int* __restrict__ cnt1, const int* __restrict__ cnt2,
            int* __restrict__ off1, int* __restrict__ off2,
            const float* __restrict__ me, float* __restrict__ laux) {
    __shared__ float red[NEXP];
    int e = threadIdx.x;
    int run = 0;
    for (int b = 0; b < NBLK; ++b) { off1[b * NEXP + e] = run; run += cnt1[b * NEXP + e]; }
    int total1 = run;
    int ce = total1 < CAP ? total1 : CAP;       // capacity-masked column sum of mask1
    int run2 = 0;
    for (int b = 0; b < NBLK; ++b) { off2[b * NEXP + e] = run2 + total1; run2 += cnt2[b * NEXP + e]; }
    red[e] = me[e] * (float)ce;
    __syncthreads();
    if (e == 0) {
        float sum = 0.f;
        for (int i = 0; i < NEXP; ++i) sum += red[i];
        laux[0] = sum * ((float)NEXP / ((float)S_TOK * (float)S_TOK));
    }
}

// ---------------------------------------------------------------------------
// K3: exact per-row locations via in-order within-block replay (thread = expert).
// ---------------------------------------------------------------------------
__global__ void __launch_bounds__(64)
locations_kernel(const int* __restrict__ idx1, const int* __restrict__ idx2,
                 const int* __restrict__ off1, const int* __restrict__ off2,
                 float* __restrict__ loc1o, float* __restrict__ loc2o) {
    __shared__ int s1[ROWS_PER_BLK], s2[ROWS_PER_BLK];
    int tid  = threadIdx.x;
    int base = blockIdx.x * ROWS_PER_BLK;
    s1[tid]      = idx1[base + tid];
    s1[tid + 64] = idx1[base + 64 + tid];
    s2[tid]      = idx2[base + tid];
    s2[tid + 64] = idx2[base + 64 + tid];
    __syncthreads();
    int e  = tid;
    int o1 = off1[blockIdx.x * NEXP + e];
    int o2 = off2[blockIdx.x * NEXP + e];
    int c1 = 0, c2 = 0;
    for (int r = 0; r < ROWS_PER_BLK; ++r) {
        if (s1[r] == e) { int l = o1 + c1; ++c1; loc1o[base + r] = (l < CAP) ? (float)l : 0.f; }
        if (s2[r] == e) { int l = o2 + c2; ++c2; loc2o[base + r] = (l < CAP) ? (float)l : 0.f; }
    }
}

// ---------------------------------------------------------------------------
extern "C" void kernel_launch(void* const* d_in, const int* in_sizes, int n_in,
                              void* d_out, int out_size, void* d_ws, size_t ws_size,
                              hipStream_t stream) {
    const float* x  = (const float*)d_in[0];   // (65536, 2048) f32
    const float* wg = (const float*)d_in[1];   // (64, 2048)    f32
    float* out = (float*)d_out;                // 1 + 6*65536 floats

    char* ws = (char*)d_ws;
    unsigned short* wgpack = (unsigned short*)(ws + 0);        // 262144 B
    float* me   = (float*)(ws + 262144);                       // 256 B
    int*   cnt1 = (int*)(ws + 262400);                         // 131072 B
    int*   cnt2 = (int*)(ws + 393472);                         // 131072 B
    int*   off1 = (int*)(ws + 524544);                         // 131072 B
    int*   off2 = (int*)(ws + 655616);                         // 131072 B
    int*   idx1 = (int*)(ws + 786688);                         // 262144 B
    int*   idx2 = (int*)(ws + 1048832);                        // 262144 B -> total ~1.25 MB

    float* laux = out;
    float* g1   = out + 1;
    float* g2   = out + 1 + 1 * S_TOK;
    float* i1f  = out + 1 + 2 * S_TOK;
    float* i2f  = out + 1 + 3 * S_TOK;
    float* l1   = out + 1 + 4 * S_TOK;
    float* l2   = out + 1 + 5 * S_TOK;

    pack_wg_kernel<<<512, 256, 0, stream>>>(wg, wgpack, me);
    gemm_top2_kernel<<<NBLK, 256, 0, stream>>>(x, wgpack, g1, g2, i1f, i2f,
                                               idx1, idx2, cnt1, cnt2, me);
    scan_kernel<<<1, 64, 0, stream>>>(cnt1, cnt2, off1, off2, me, laux);
    locations_kernel<<<NBLK, 64, 0, stream>>>(idx1, idx2, off1, off2, l1, l2);
}
